// NeuralGrangerCausality_6966436954287
// MI455X (gfx1250) — matrix operand order repaired and checked by reading, not compiled, not run on hardware
//
#include <hip/hip_runtime.h>
#include <hip/hip_bf16.h>
#include <math.h>

// Problem constants (match reference)
#define T_DIM   64
#define N_NODES 2000
#define F_INPUT 32
#define H_DIM   64
#define C_CLS   10
#define E_EDGES 32000
#define EPLUSN  (E_EDGES + N_NODES)   // 34000
#define K_PAD   2016                  // 2000 padded to multiple of 32
#define BN_EPS  1e-5f

typedef __attribute__((ext_vector_type(16))) _Float16 v16h;
typedef __attribute__((ext_vector_type(8)))  float    v8f;
typedef __attribute__((ext_vector_type(4)))  unsigned int v4u;
typedef __attribute__((ext_vector_type(4)))  int v4i;
typedef __attribute__((ext_vector_type(8)))  int v8i;

#if defined(__has_builtin)
#if __has_builtin(__builtin_amdgcn_tensor_load_to_lds) && __has_builtin(__builtin_amdgcn_s_wait_tensorcnt)
#define CAUSAL_USE_TDM 1
#endif
#endif
#ifndef CAUSAL_USE_TDM
#define CAUSAL_USE_TDM 0
#endif

__device__ __forceinline__ float sigmoidf_(float x) { return 1.0f / (1.0f + __expf(-x)); }

// ---------------------------------------------------------------------------
// Shared WMMA fragment micro-kernel: one 16x16x32 MAC step.
// A from global row (interleaved half-wave K layout), B from an LDS tile
// laid out row-major [32][H_DIM] f16.
// ---------------------------------------------------------------------------
__device__ __forceinline__ v8f mma_chunk_lds(const _Float16* __restrict__ arow, int kA,
                                             const _Float16* __restrict__ btile, int kB,
                                             int nt, v8f acc)
{
    v16h a, b;
#pragma unroll
    for (int i = 0; i < 8; ++i) {
        const int kk = kA + ((i < 4) ? 2 * i : 16 + 2 * (i - 4));
        a[2 * i]     = arow[kk];
        a[2 * i + 1] = arow[kk + 1];
    }
    const _Float16* br = btile + nt * 16 + (kB & 15);
    const int kb0 = (kB >> 4) * 8;
#pragma unroll
    for (int j = 0; j < 8; ++j) {
        const int kk = kb0 + ((j < 4) ? 2 * j : 16 + 2 * (j - 4));
        b[2 * j]     = br[kk * H_DIM];
        b[2 * j + 1] = br[(kk + 1) * H_DIM];
    }
    return __builtin_amdgcn_wmma_f32_16x16x32_f16(false, a, false, b, (short)0, acc, false, false);
}

#if CAUSAL_USE_TDM
// ---------------------------------------------------------------------------
// Issue one TDM transfer: 32 rows x 64 cols of f16 (4 KB) global -> LDS.
// D# built per CDNA5 ISA 8.3/8.4: 2D tile, data_size=2B,
// tile_dim0=64, tile_dim1=32, tensor_dim0_stride=64.
// ---------------------------------------------------------------------------
__device__ __forceinline__ void tdm_load_btile(const _Float16* gsrc, _Float16* ldst)
{
    const unsigned long long ga  = (unsigned long long)(uintptr_t)gsrc;
    const unsigned int       lds = (unsigned int)(uintptr_t)ldst;   // LDS offset = addr[31:0]

    v4u g0;
    g0[0] = 1u;                                                    // count=1, user mode
    g0[1] = lds;                                                   // lds_addr
    g0[2] = (unsigned int)ga;                                      // global_addr[31:0]
    g0[3] = (unsigned int)((ga >> 32) & 0x01FFFFFFu) | (2u << 30); // ga[56:32] | type=2

    v8i g1;
    g1[0] = 0x00010000;        // data_size=1 (2 bytes); workgroup_mask=0
    g1[1] = (int)(64u << 16);  // tensor_dim0 = 64       (bits 63:48)
    g1[2] = (int)(32u << 16);  // tensor_dim1 = 32       (bits 95:80)
    g1[3] = (int)(64u << 16);  // tile_dim0   = 64       (bits 127:112)
    g1[4] = 32;                // tile_dim1   = 32       (bits 143:128)
    g1[5] = 64;                // tensor_dim0_stride = 64 (bits 191:160)
    g1[6] = 0;
    g1[7] = 0;

    v4i gz = {0, 0, 0, 0};
#if __clang_major__ >= 23
    v8i gz8 = {0, 0, 0, 0, 0, 0, 0, 0};
    __builtin_amdgcn_tensor_load_to_lds(g0, g1, gz, gz, gz8, 0);
#else
    __builtin_amdgcn_tensor_load_to_lds(g0, g1, gz, gz, 0);
#endif
}
#endif

// ---------------------------------------------------------------------------
// Causal-aggregation GEMM: C[t] = wc[2000 x K_PAD] * x[t][K_PAD x 64].
// grid(125, 1, T), block(32,4): 4 waves share one 16-row M tile; wave w owns
// N tile w. B K-chunks staged into LDS by the Tensor Data Mover
// (double-buffered, s_wait_tensorcnt), fragments consumed via ds_load.
// ---------------------------------------------------------------------------
__global__ void causal_gemm_kernel(const _Float16* __restrict__ A,   // wc  [N][K_PAD]
                                   const _Float16* __restrict__ B,   // xh  [T][K_PAD][H]
                                   float* __restrict__ C)            // out [T][N][H]
{
    __shared__ _Float16 btile[2][32 * H_DIM];   // 2 x 4 KB
    const int lane = threadIdx.x;               // 0..31
    const int wv   = threadIdx.y;               // 0..3  == N tile
    const int mt   = blockIdx.x;
    const int t    = blockIdx.z;

    const _Float16* Bt   = B + (long long)t * K_PAD * H_DIM;
    const _Float16* Arow = A + (long long)(mt * 16 + (lane & 15)) * K_PAD;
    const int kA = (lane >> 4) * 8;                    // A half-wave K offset
    const int kB = (lane & 15) + (lane >> 4) * 16;     // B lane -> column (low 4 bits), K half (bit 4)

    v8f acc = {};

#if CAUSAL_USE_TDM
    if (wv == 0) tdm_load_btile(Bt, &btile[0][0]);
    for (int kc = 0; kc < K_PAD; kc += 32) {
        const int buf = (kc >> 5) & 1;
        if (wv == 0) {
            if (kc + 32 < K_PAD) {
                tdm_load_btile(Bt + (long long)(kc + 32) * H_DIM, &btile[buf ^ 1][0]);
                __builtin_amdgcn_s_wait_tensorcnt(1);   // current buffer complete
            } else {
                __builtin_amdgcn_s_wait_tensorcnt(0);
            }
        }
        __syncthreads();
        __builtin_prefetch(Arow + kc + 32, 0, 1);
        acc = mma_chunk_lds(Arow + kc, kA, &btile[buf][0], kB, wv, acc);
        __syncthreads();   // protect buffer before next TDM overwrite
    }
#else
    const int lt = wv * 32 + lane;   // 0..127 cooperative loader
    for (int kc = 0; kc < K_PAD; kc += 32) {
        for (int i = lt; i < 32 * H_DIM; i += 128)
            btile[0][i] = Bt[(long long)kc * H_DIM + i];
        __syncthreads();
        __builtin_prefetch(Arow + kc + 32, 0, 1);
        acc = mma_chunk_lds(Arow + kc, kA, &btile[0][0], kB, wv, acc);
        __syncthreads();
    }
#endif

    const int crow = mt * 16 + (lane >> 4) * 8;
    const int ccol = wv * 16 + (lane & 15);
    float* Ct = C + (long long)t * N_NODES * H_DIM;
#pragma unroll
    for (int r = 0; r < 8; ++r)
        Ct[(long long)(crow + r) * H_DIM + ccol] = acc[r];
}

// ---------------------------------------------------------------------------
// Generic WMMA GEMM: C[M x Ncols] = A[M x K] * B[K x Ncols], f16 in, f32 out.
// One wave per 16x16 tile; K multiple of 32; M, Ncols multiples of 16.
// ---------------------------------------------------------------------------
__global__ void wmma_gemm_f16f32(const _Float16* __restrict__ A,
                                 const _Float16* __restrict__ B,
                                 float* __restrict__ C,
                                 int K, int Ncols, int lda, int ldb, int ldc,
                                 long long strideA, long long strideB, long long strideC)
{
    const int lane = threadIdx.x;
    const int mt   = blockIdx.x;
    const int nt   = blockIdx.y * blockDim.y + threadIdx.y;
    if (nt * 16 >= Ncols) return;   // wave-uniform, never divergent in our launches

    const _Float16* Ab = A + (long long)blockIdx.z * strideA;
    const _Float16* Bb = B + (long long)blockIdx.z * strideB;
    float*          Cb = C + (long long)blockIdx.z * strideC;

    const int mrow = mt * 16 + (lane & 15);
    const int kA   = (lane >> 4) * 8;
    const int kB   = (lane & 15) + (lane >> 4) * 16;

    v8f acc = {};
    for (int kc = 0; kc < K; kc += 32) {
        const _Float16* arow = Ab + (long long)mrow * lda + kc;
        const _Float16* brow = Bb + (long long)kc * ldb + nt * 16 + (lane & 15);
        __builtin_prefetch(arow + 32, 0, 1);

        v16h a, b;
#pragma unroll
        for (int i = 0; i < 8; ++i) {
            const int kk = kA + ((i < 4) ? 2 * i : 16 + 2 * (i - 4));
            a[2 * i]     = arow[kk];
            a[2 * i + 1] = arow[kk + 1];
        }
#pragma unroll
        for (int j = 0; j < 8; ++j) {
            const int kk = kA + ((j < 4) ? 2 * j : 16 + 2 * (j - 4));
            b[2 * j]     = brow[(long long)kk * ldb];
            b[2 * j + 1] = brow[(long long)(kk + 1) * ldb];
        }
        acc = __builtin_amdgcn_wmma_f32_16x16x32_f16(
                  false, a, false, b, (short)0, acc, false, false);
    }

    const int crow = mt * 16 + (lane >> 4) * 8;
    const int ccol = nt * 16 + (lane & 15);
#pragma unroll
    for (int r = 0; r < 8; ++r)
        Cb[(long long)(crow + r) * ldc + ccol] = acc[r];
}

// ---------------------------------------------------------------------------
// Row softmax of causal_weight [N,N] -> f16 [N, K_PAD] (cols >= N zeroed)
// ---------------------------------------------------------------------------
__global__ void softmax_rows_kernel(const float* __restrict__ cw, _Float16* __restrict__ out)
{
    __shared__ float red[256];
    const int row = blockIdx.x;
    const int tid = threadIdx.x;
    const float* r = cw + (long long)row * N_NODES;

    float lm = -1e30f;
    for (int c = tid; c < N_NODES; c += 256) lm = fmaxf(lm, r[c]);
    red[tid] = lm; __syncthreads();
    for (int s = 128; s > 0; s >>= 1) { if (tid < s) red[tid] = fmaxf(red[tid], red[tid + s]); __syncthreads(); }
    const float m = red[0]; __syncthreads();

    float ls = 0.f;
    for (int c = tid; c < N_NODES; c += 256) ls += __expf(r[c] - m);
    red[tid] = ls; __syncthreads();
    for (int s = 128; s > 0; s >>= 1) { if (tid < s) red[tid] += red[tid + s]; __syncthreads(); }
    const float inv = 1.0f / red[0];

    _Float16* o = out + (long long)row * K_PAD;
    for (int c = tid; c < K_PAD; c += 256)
        o[c] = (c < N_NODES) ? (_Float16)(__expf(r[c] - m) * inv) : (_Float16)0.f;
}

// ---------------------------------------------------------------------------
// lin_in + relu -> x_h f16, layout [T][K_PAD][H] (pad rows zero for WMMA K)
// ---------------------------------------------------------------------------
__global__ void lin_in_kernel(const float* __restrict__ xseq,
                              const float* __restrict__ w,   // [H, F_INPUT]
                              const float* __restrict__ b,
                              _Float16* __restrict__ xh)
{
    const int t = blockIdx.x / K_PAD;
    const int j = blockIdx.x % K_PAD;
    const int h = threadIdx.x;
    _Float16* o = xh + ((long long)t * K_PAD + j) * H_DIM;
    if (j >= N_NODES) { o[h] = (_Float16)0.f; return; }

    __shared__ float xr[F_INPUT];
    if (h < F_INPUT) xr[h] = xseq[((long long)t * N_NODES + j) * F_INPUT + h];
    __syncthreads();

    float s = b[h];
#pragma unroll
    for (int f = 0; f < F_INPUT; ++f) s += xr[f] * w[h * F_INPUT + f];
    o[h] = (_Float16)fmaxf(s, 0.f);
}

// ---------------------------------------------------------------------------
// fusion: relu([x, x_agg] @ fusion_w.T + b) -> x2_h f16 [T*N][H]
// ---------------------------------------------------------------------------
__global__ void fusion_kernel(const _Float16* __restrict__ xh,   // [T][K_PAD][H]
                              const float* __restrict__ xagg,    // [T][N][H]
                              const float* __restrict__ fw,      // [H, 2H]
                              const float* __restrict__ fb,
                              _Float16* __restrict__ x2h)
{
    const int t = blockIdx.x / N_NODES;
    const int n = blockIdx.x % N_NODES;
    const int h = threadIdx.x;
    __shared__ float xr[2 * H_DIM];
    xr[h]         = (float)xh[((long long)t * K_PAD + n) * H_DIM + h];
    xr[H_DIM + h] = xagg[((long long)t * N_NODES + n) * H_DIM + h];
    __syncthreads();

    float s = fb[h];
#pragma unroll 8
    for (int k = 0; k < 2 * H_DIM; ++k) s += xr[k] * fw[h * 2 * H_DIM + k];
    x2h[((long long)t * N_NODES + n) * H_DIM + h] = (_Float16)fmaxf(s, 0.f);
}

// ---------------------------------------------------------------------------
__global__ void fill_f32_kernel(float* p, long long n, float v)
{
    long long i = (long long)blockIdx.x * blockDim.x + threadIdx.x;
    if (i < n) p[i] = v;
}

__global__ void transpose_convert_kernel(const float* __restrict__ W, _Float16* __restrict__ Wt,
                                         int R, int Cc)
{
    int i = blockIdx.x * blockDim.x + threadIdx.x;
    if (i >= R * Cc) return;
    int r = i / Cc, c = i % Cc;
    Wt[(long long)c * R + r] = (_Float16)W[i];   // Wt[k][n] = W[n][k]
}

__global__ void deg_kernel(const int* __restrict__ ei, const float* __restrict__ ew,
                           float* __restrict__ deg)
{
    int e = blockIdx.x * blockDim.x + threadIdx.x;
    if (e >= EPLUSN) return;
    int dst; float w;
    if (e < E_EDGES) { dst = ei[E_EDGES + e]; w = ew[e]; }
    else             { dst = e - E_EDGES;     w = 1.0f;  }
    atomicAdd(&deg[dst], w);
}

__global__ void norm_kernel(const int* __restrict__ ei, const float* __restrict__ ew,
                            const float* __restrict__ deg, float* __restrict__ norm)
{
    int e = blockIdx.x * blockDim.x + threadIdx.x;
    if (e >= EPLUSN) return;
    int src, dst; float w;
    if (e < E_EDGES) { src = ei[e]; dst = ei[E_EDGES + e]; w = ew[e]; }
    else             { src = dst = e - E_EDGES;            w = 1.0f;  }
    float ds = rsqrtf(fmaxf(deg[src], 1e-12f));
    float dd = rsqrtf(fmaxf(deg[dst], 1e-12f));
    norm[e] = ds * w * dd;
}

__global__ void scatter_kernel(const float* __restrict__ xw, const int* __restrict__ ei,
                               const float* __restrict__ norm, float* __restrict__ agg)
{
    const int e = blockIdx.x;
    const int t = blockIdx.y;
    const int h = threadIdx.x;
    int src, dst;
    if (e < E_EDGES) { src = ei[e]; dst = ei[E_EDGES + e]; }
    else             { src = dst = e - E_EDGES; }
    const float msg = norm[e] * xw[((long long)t * N_NODES + src) * H_DIM + h];
    atomicAdd(&agg[((long long)t * N_NODES + dst) * H_DIM + h], msg);
}

__global__ void bias_bn_relu_f16_kernel(const float* __restrict__ agg,
                                        const float* __restrict__ bias,
                                        const float* __restrict__ g, const float* __restrict__ b,
                                        const float* __restrict__ m, const float* __restrict__ v,
                                        _Float16* __restrict__ out, long long total)
{
    long long i = (long long)blockIdx.x * blockDim.x + threadIdx.x;
    if (i >= total) return;
    int h = (int)(i & (H_DIM - 1));
    float val = (agg[i] + bias[h] - m[h]) * rsqrtf(v[h] + BN_EPS) * g[h] + b[h];
    out[i] = (_Float16)fmaxf(val, 0.f);
}

__global__ void gru_gate_kernel(const float* __restrict__ gi, const float* __restrict__ gh,
                                const float* __restrict__ bih, const float* __restrict__ bhh,
                                float* __restrict__ hf, _Float16* __restrict__ hh,
                                _Float16* __restrict__ ysh, int t, int writeYs)
{
    int i = blockIdx.x * blockDim.x + threadIdx.x;
    if (i >= N_NODES * H_DIM) return;
    const int n = i >> 6, h = i & 63;
    const long long base = (long long)n * 3 * H_DIM;
    float r  = sigmoidf_(gi[base + h]           + bih[h]           + gh[base + h]           + bhh[h]);
    float z  = sigmoidf_(gi[base + H_DIM + h]   + bih[H_DIM + h]   + gh[base + H_DIM + h]   + bhh[H_DIM + h]);
    float nn = tanhf    (gi[base + 2*H_DIM + h] + bih[2*H_DIM + h] + r * (gh[base + 2*H_DIM + h] + bhh[2*H_DIM + h]));
    float hp = hf[i];
    float hn = (1.0f - z) * nn + z * hp;
    hf[i] = hn;
    hh[i] = (_Float16)hn;
    if (writeYs) ysh[(long long)t * N_NODES * H_DIM + i] = (_Float16)hn;
}

__global__ void head_kernel(const float* __restrict__ hf,
                            const float* __restrict__ g, const float* __restrict__ b,
                            const float* __restrict__ m, const float* __restrict__ v,
                            const float* __restrict__ lw, const float* __restrict__ lb,
                            float* __restrict__ out)
{
    const int n = blockIdx.x, h = threadIdx.x;
    __shared__ float sh[H_DIM];
    __shared__ float logit[C_CLS];
    __shared__ float mx, lse;
    float val = (hf[(long long)n * H_DIM + h] - m[h]) * rsqrtf(v[h] + BN_EPS) * g[h] + b[h];
    sh[h] = fmaxf(val, 0.f);
    __syncthreads();
    if (h < C_CLS) {
        float s = lb[h];
#pragma unroll
        for (int k = 0; k < H_DIM; ++k) s += sh[k] * lw[h * H_DIM + k];
        logit[h] = s;
    }
    __syncthreads();
    if (h == 0) {
        float mm = -1e30f;
        for (int c = 0; c < C_CLS; ++c) mm = fmaxf(mm, logit[c]);
        float s = 0.f;
        for (int c = 0; c < C_CLS; ++c) s += __expf(logit[c] - mm);
        mx = mm; lse = logf(s);
    }
    __syncthreads();
    if (h < C_CLS) out[(long long)n * C_CLS + h] = logit[h] - mx - lse;
}

// ---------------------------------------------------------------------------
extern "C" void kernel_launch(void* const* d_in, const int* in_sizes, int n_in,
                              void* d_out, int out_size, void* d_ws, size_t ws_size,
                              hipStream_t stream)
{
    const float* x_seq      = (const float*)d_in[0];
    const int*   edge_index = (const int*)  d_in[1];
    const float* edge_w     = (const float*)d_in[2];
    const float* causal_w   = (const float*)d_in[3];
    const float* lin_in_w   = (const float*)d_in[4];
    const float* lin_in_b   = (const float*)d_in[5];
    const float* fusion_w   = (const float*)d_in[6];
    const float* fusion_b   = (const float*)d_in[7];
    const float* gcn_w0     = (const float*)d_in[8];
    const float* gcn_b0     = (const float*)d_in[9];
    const float* gcn_w1     = (const float*)d_in[10];
    const float* gcn_b1     = (const float*)d_in[11];
    const float* bn0_g = (const float*)d_in[12], *bn0_b = (const float*)d_in[13];
    const float* bn0_m = (const float*)d_in[14], *bn0_v = (const float*)d_in[15];
    const float* bn1_g = (const float*)d_in[16], *bn1_b = (const float*)d_in[17];
    const float* bn1_m = (const float*)d_in[18], *bn1_v = (const float*)d_in[19];
    const float* wih0 = (const float*)d_in[20], *whh0 = (const float*)d_in[21];
    const float* bih0 = (const float*)d_in[22], *bhh0 = (const float*)d_in[23];
    const float* wih1 = (const float*)d_in[24], *whh1 = (const float*)d_in[25];
    const float* bih1 = (const float*)d_in[26], *bhh1 = (const float*)d_in[27];
    const float* bno_g = (const float*)d_in[28], *bno_b = (const float*)d_in[29];
    const float* bno_m = (const float*)d_in[30], *bno_v = (const float*)d_in[31];
    const float* lout_w = (const float*)d_in[32], *lout_b = (const float*)d_in[33];
    float* out = (float*)d_out;

    // ---- workspace carve-out (256B aligned regions) ----
    char* p = (char*)d_ws;
    auto carve = [&](size_t bytes) -> void* {
        void* r = (void*)p;
        p += (bytes + 255) & ~(size_t)255;
        return r;
    };
    const long long TNH = (long long)T_DIM * N_NODES * H_DIM;   // 8,192,000
    _Float16* wch  = (_Float16*)carve((size_t)N_NODES * K_PAD * 2);
    _Float16* xh   = (_Float16*)carve((size_t)T_DIM * K_PAD * H_DIM * 2);
    float*    xagg = (float*)   carve((size_t)TNH * 4);
    _Float16* x2h  = (_Float16*)carve((size_t)TNH * 2);
    float*    xw   = (float*)   carve((size_t)TNH * 4);
    float*    agg  = (float*)   carve((size_t)TNH * 4);
    _Float16* x3h  = (_Float16*)carve((size_t)TNH * 2);
    _Float16* x4h  = (_Float16*)carve((size_t)TNH * 2);
    _Float16* ysh  = (_Float16*)carve((size_t)TNH * 2);
    float*    gi   = (float*)   carve((size_t)N_NODES * 3 * H_DIM * 4);
    float*    gh   = (float*)   carve((size_t)N_NODES * 3 * H_DIM * 4);
    float*    hf   = (float*)   carve((size_t)N_NODES * H_DIM * 4);
    _Float16* hh   = (_Float16*)carve((size_t)N_NODES * H_DIM * 2);
    float*    deg  = (float*)   carve((size_t)N_NODES * 4);
    float*    nrm  = (float*)   carve((size_t)EPLUSN * 4);
    _Float16* w0t  = (_Float16*)carve((size_t)H_DIM * H_DIM * 2);
    _Float16* w1t  = (_Float16*)carve((size_t)H_DIM * H_DIM * 2);
    _Float16* wih0t= (_Float16*)carve((size_t)H_DIM * 3 * H_DIM * 2);
    _Float16* whh0t= (_Float16*)carve((size_t)H_DIM * 3 * H_DIM * 2);
    _Float16* wih1t= (_Float16*)carve((size_t)H_DIM * 3 * H_DIM * 2);
    _Float16* whh1t= (_Float16*)carve((size_t)H_DIM * 3 * H_DIM * 2);

    const dim3 wmmaBlk(32, 4);

    // ---- weight conversions (f32 -> transposed f16 for WMMA B operands) ----
    transpose_convert_kernel<<<(H_DIM*H_DIM + 255)/256, 256, 0, stream>>>(gcn_w0, w0t, H_DIM, H_DIM);
    transpose_convert_kernel<<<(H_DIM*H_DIM + 255)/256, 256, 0, stream>>>(gcn_w1, w1t, H_DIM, H_DIM);
    transpose_convert_kernel<<<(3*H_DIM*H_DIM + 255)/256, 256, 0, stream>>>(wih0, wih0t, 3*H_DIM, H_DIM);
    transpose_convert_kernel<<<(3*H_DIM*H_DIM + 255)/256, 256, 0, stream>>>(whh0, whh0t, 3*H_DIM, H_DIM);
    transpose_convert_kernel<<<(3*H_DIM*H_DIM + 255)/256, 256, 0, stream>>>(wih1, wih1t, 3*H_DIM, H_DIM);
    transpose_convert_kernel<<<(3*H_DIM*H_DIM + 255)/256, 256, 0, stream>>>(whh1, whh1t, 3*H_DIM, H_DIM);

    // ---- GCN normalization coefficients ----
    fill_f32_kernel<<<(N_NODES + 255)/256, 256, 0, stream>>>(deg, N_NODES, 0.f);
    deg_kernel<<<(EPLUSN + 255)/256, 256, 0, stream>>>(edge_index, edge_w, deg);
    norm_kernel<<<(EPLUSN + 255)/256, 256, 0, stream>>>(edge_index, edge_w, deg, nrm);

    // ---- front end ----
    softmax_rows_kernel<<<N_NODES, 256, 0, stream>>>(causal_w, wch);
    lin_in_kernel<<<T_DIM * K_PAD, H_DIM, 0, stream>>>(x_seq, lin_in_w, lin_in_b, xh);

    // x_agg[t] = wc @ x[t] : TDM-staged B, 4 waves/block
    causal_gemm_kernel<<<dim3(N_NODES/16, 1, T_DIM), wmmaBlk, 0, stream>>>(wch, xh, xagg);

    fusion_kernel<<<T_DIM * N_NODES, H_DIM, 0, stream>>>(xh, xagg, fusion_w, fusion_b, x2h);

    // ---- GCN layer 0 ----
    wmma_gemm_f16f32<<<dim3((T_DIM*N_NODES)/16, 1, 1), wmmaBlk, 0, stream>>>(
        x2h, w0t, xw, H_DIM, H_DIM, H_DIM, H_DIM, H_DIM, 0LL, 0LL, 0LL);
    fill_f32_kernel<<<(int)((TNH + 255)/256), 256, 0, stream>>>(agg, TNH, 0.f);
    scatter_kernel<<<dim3(EPLUSN, T_DIM), H_DIM, 0, stream>>>(xw, edge_index, nrm, agg);
    bias_bn_relu_f16_kernel<<<(int)((TNH + 255)/256), 256, 0, stream>>>(
        agg, gcn_b0, bn0_g, bn0_b, bn0_m, bn0_v, x3h, TNH);

    // ---- GCN layer 1 ----
    wmma_gemm_f16f32<<<dim3((T_DIM*N_NODES)/16, 1, 1), wmmaBlk, 0, stream>>>(
        x3h, w1t, xw, H_DIM, H_DIM, H_DIM, H_DIM, H_DIM, 0LL, 0LL, 0LL);
    fill_f32_kernel<<<(int)((TNH + 255)/256), 256, 0, stream>>>(agg, TNH, 0.f);
    scatter_kernel<<<dim3(EPLUSN, T_DIM), H_DIM, 0, stream>>>(xw, edge_index, nrm, agg);
    bias_bn_relu_f16_kernel<<<(int)((TNH + 255)/256), 256, 0, stream>>>(
        agg, gcn_b1, bn1_g, bn1_b, bn1_m, bn1_v, x4h, TNH);

    // ---- GRU layers (sequential scan; WMMA for gi and gh matmuls) ----
    const int NH = N_NODES * H_DIM;
    for (int layer = 0; layer < 2; ++layer) {
        const _Float16* xin  = (layer == 0) ? x4h  : ysh;
        const _Float16* wiht = (layer == 0) ? wih0t : wih1t;
        const _Float16* whht = (layer == 0) ? whh0t : whh1t;
        const float*    bih  = (layer == 0) ? bih0 : bih1;
        const float*    bhh  = (layer == 0) ? bhh0 : bhh1;
        fill_f32_kernel<<<(NH + 255)/256, 256, 0, stream>>>(hf, NH, 0.f);
        fill_f32_kernel<<<(NH/2 + 255)/256, 256, 0, stream>>>((float*)hh, NH/2, 0.f);

        for (int t = 0; t < T_DIM; ++t) {
            wmma_gemm_f16f32<<<dim3(N_NODES/16, 3, 1), wmmaBlk, 0, stream>>>(
                xin + (long long)t * NH, wiht, gi,
                H_DIM, 3*H_DIM, H_DIM, 3*H_DIM, 3*H_DIM, 0LL, 0LL, 0LL);
            wmma_gemm_f16f32<<<dim3(N_NODES/16, 3, 1), wmmaBlk, 0, stream>>>(
                hh, whht, gh,
                H_DIM, 3*H_DIM, H_DIM, 3*H_DIM, 3*H_DIM, 0LL, 0LL, 0LL);
            gru_gate_kernel<<<(NH + 255)/256, 256, 0, stream>>>(
                gi, gh, bih, bhh, hf, hh, ysh, t, (layer == 0) ? 1 : 0);
        }
    }

    // ---- head ----
    head_kernel<<<N_NODES, H_DIM, 0, stream>>>(
        hf, bno_g, bno_b, bno_m, bno_v, lout_w, lout_b, out);
}
